// MMDLoss_5634997093232
// MI455X (gfx1250) — matrix operand order repaired and checked by reading, not compiled, not run on hardware
//
#include <hip/hip_runtime.h>
#include <hip/hip_bf16.h>

// ---------------------------------------------------------------------------
// MMD loss (Gaussian kernel, sigma=1) for N=8192, D=128 fp32 inputs.
//
// mmd = scale*(sum_offdiag Kxx) + scale*(sum_offdiag Kyy) - 2*sum(Kxy)/N^2
//   K(a,b) = exp(-(||a||^2 + ||b||^2 - 2 a.b)/2),  scale = (N-1)/N
//
// MI455X / gfx1250 strategy (wave32, WMMA):
//  * fp32 split into bf16 hi + bf16 lo; dot = hi.hi + hi.lo + lo.hi via
//    3x v_wmma_f32_16x16x32_bf16 per 16x16x32 step (~fp32 accurate).
//  * 128x128 superblock per 256-thread workgroup (8 waves, 4x2 wave grid,
//    each wave owns 2x4 16x16 tiles).
//  * K staged through *double-buffered* LDS in chunks of 32 using
//    GLOBAL_LOAD_ASYNC_TO_LDS_B128 (ASYNCcnt) so the DMA of chunk k+1
//    overlaps the WMMA work on chunk k; one s_wait_asynccnt + barrier per
//    chunk. Falls back to sync global->VGPR->ds_store if the async builtin
//    is not declared by this toolchain.
//  * epilogue fuses d2 -> clamp -> v_exp_f32 -> sum; diagonal excluded
//    exactly; symmetric matrices compute upper triangle only (weight 2).
//  * block partial -> LDS tree -> global_atomic_add_f64 into ws accumulators.
// ---------------------------------------------------------------------------

typedef __attribute__((ext_vector_type(16))) __bf16 v16bf;
typedef __attribute__((ext_vector_type(8)))  float  v8f;
typedef int v4i_g __attribute__((vector_size(16)));   // matches builtin param type

static_assert(sizeof(v16bf) == 32, "v16bf must be 32 bytes");

#define KD   128   // feature dim (fixed by reference)
#define BLK  128   // superblock edge
#define KC   32    // K chunk staged in LDS (one WMMA K-step)
#define NK   (KD / KC)
#define LSTR 40    // LDS row stride in bf16 (32 data + 8 pad = 80B, 16B aligned)

// log2(e)/2 : exp(-d2/2) == exp2(-d2 * 0.72134752...)
#define NEG_HALF_LOG2E 0.72134752044448170f

#define AS1 __attribute__((address_space(1)))
#define AS3 __attribute__((address_space(3)))

#if defined(__has_builtin)
#if __has_builtin(__builtin_amdgcn_global_load_async_to_lds_b128)
#define HAVE_ASYNC_LDS 1
#endif
#endif
#ifndef HAVE_ASYNC_LDS
#define HAVE_ASYNC_LDS 0
#endif

__device__ __forceinline__ void wait_async_all() {
#if HAVE_ASYNC_LDS
#if __has_builtin(__builtin_amdgcn_s_wait_asynccnt)
  __builtin_amdgcn_s_wait_asynccnt(0);
#else
  asm volatile("s_wait_asynccnt 0" ::: "memory");
#endif
#endif
}

#if HAVE_ASYNC_LDS
// 16B/lane async DMA global -> LDS, tracked by ASYNCcnt.
// Builtin prototype (from compiler diagnostic): (v4i AS1*, v4i AS3*, imm, imm).
__device__ __forceinline__ void async_copy16(const __bf16* gp, __bf16* lp) {
  __builtin_amdgcn_global_load_async_to_lds_b128(
      (AS1 v4i_g*)(AS1 void*)gp, (AS3 v4i_g*)(AS3 void*)lp,
      /*offset=*/0, /*cpol=*/0);
}
#endif

// ---------------------------------------------------------------------------
// Fragment loaders from LDS (row-major, LSTR-bf16 row stride).
//
// A-matrix 16x32 bf16 (ISA 7.12.2): lane l holds row (l&15); with h = l>>4,
// vector elements 0..7  = K[8h .. 8h+7]      (16B contiguous run)
// vector elements 8..15 = K[16+8h .. 23+8h]  (16B contiguous run)
__device__ __forceinline__ v16bf load_a_frag(const __bf16* row, int h) {
  union { v16bf v; uint4 q[2]; } u;
  u.q[0] = *reinterpret_cast<const uint4*>(row + 8 * h);
  u.q[1] = *reinterpret_cast<const uint4*>(row + 16 + 8 * h);
  return u.v;
}
// B-matrix 32x16 bf16 (dense analogue of ISA 7.12.4 sparse-B blocks):
// lane l holds column (l&15); elements 0..7 = K[8h .. 8h+7], 8..15 = K[16+8h .. 23+8h].
__device__ __forceinline__ v16bf load_b_frag(const __bf16* row, int h) {
  union { v16bf v; uint4 q[2]; } u;
  u.q[0] = *reinterpret_cast<const uint4*>(row + 8 * h);
  u.q[1] = *reinterpret_cast<const uint4*>(row + 16 + 8 * h);
  return u.v;
}

// ---------------------------------------------------------------------------
// Stage one K-chunk (4 arrays x 128 rows x 32 bf16) into an LDS buffer.
// 2048 16B segments over 256 threads = 8 per thread; the array select (i>>1)
// is a compile-time constant after unrolling -> branch-free address math.
__device__ __forceinline__ void stage_chunk(
    const __bf16* __restrict__ Ahi, const __bf16* __restrict__ Alo,
    const __bf16* __restrict__ Bhi, const __bf16* __restrict__ Blo,
    __bf16* dA0, __bf16* dA1, __bf16* dB0, __bf16* dB1,
    int mb0, int nb0, int kc, int tid)
{
#pragma unroll
  for (int i = 0; i < 8; ++i) {
    const int arr = i >> 1;                      // 0 Ahi, 1 Alo, 2 Bhi, 3 Blo
    const int rem = ((i & 1) << 8) + tid;        // 0..511
    const int row = rem >> 2;
    const int seg = rem & 3;
    const __bf16* g = (arr == 0) ? Ahi : (arr == 1) ? Alo : (arr == 2) ? Bhi : Blo;
    __bf16*      d = (arr == 0) ? dA0 : (arr == 1) ? dA1 : (arr == 2) ? dB0 : dB1;
    const int gr = ((arr < 2) ? mb0 : nb0) + row;
    const __bf16* gp = g + (size_t)gr * KD + kc * KC + seg * 8;
    __bf16*       lp = d + row * LSTR + seg * 8;
#if HAVE_ASYNC_LDS
    async_copy16(gp, lp);
#else
    if (kc + 1 < NK) __builtin_prefetch(gp + KC, 0, 1);  // global_prefetch_b8
    *reinterpret_cast<uint4*>(lp) = *reinterpret_cast<const uint4*>(gp);
#endif
  }
}

// ---------------------------------------------------------------------------
// One K-chunk of WMMA work for this wave: 2 M-tiles x 4 N-tiles x 3 products.
__device__ __forceinline__ void compute_chunk(
    const __bf16* a0, const __bf16* a1, const __bf16* b0, const __bf16* b1,
    int wm, int wn, int lm, int h, v8f acc[2][4])
{
  v16bf ah[2], al[2];
#pragma unroll
  for (int mt = 0; mt < 2; ++mt) {
    const int r = (wm * 32 + mt * 16 + lm) * LSTR;
    ah[mt] = load_a_frag(a0 + r, h);
    al[mt] = load_a_frag(a1 + r, h);
  }
#pragma unroll
  for (int nt = 0; nt < 4; ++nt) {
    const int r = (wn * 64 + nt * 16 + lm) * LSTR;
    const v16bf bh = load_b_frag(b0 + r, h);
    const v16bf bl = load_b_frag(b1 + r, h);
#pragma unroll
    for (int mt = 0; mt < 2; ++mt) {
      // dot ~= hi.hi + hi.lo + lo.hi  (lo.lo term ~2^-16, dropped)
      acc[mt][nt] = __builtin_amdgcn_wmma_f32_16x16x32_bf16(
          false, ah[mt], false, bh, (short)0, acc[mt][nt], false, false);
      acc[mt][nt] = __builtin_amdgcn_wmma_f32_16x16x32_bf16(
          false, ah[mt], false, bl, (short)0, acc[mt][nt], false, false);
      acc[mt][nt] = __builtin_amdgcn_wmma_f32_16x16x32_bf16(
          false, al[mt], false, bh, (short)0, acc[mt][nt], false, false);
    }
  }
}

// ---------------------------------------------------------------------------
// Prep: bf16 hi/lo split + exact fp32 row norms; also zeroes accumulators.
__global__ __launch_bounds__(128) void mmd_prep_kernel(
    const float* __restrict__ X, const float* __restrict__ Y,
    __bf16* __restrict__ xhi, __bf16* __restrict__ xlo,
    __bf16* __restrict__ yhi, __bf16* __restrict__ ylo,
    float* __restrict__ xn, float* __restrict__ yn,
    double* __restrict__ accs)
{
  const int row = blockIdx.x;
  const int which = blockIdx.y;     // 0 -> z, 1 -> z_prior
  const int t = threadIdx.x;

  if (row == 0 && which == 0 && t < 3) accs[t] = 0.0;  // before mains (stream order)

  const float* src = which ? Y : X;
  __bf16* hi = which ? yhi : xhi;
  __bf16* lo = which ? ylo : xlo;
  float*  nrm = which ? yn : xn;

  const float v = src[(size_t)row * KD + t];
  const __bf16 h = (__bf16)v;
  const __bf16 l = (__bf16)(v - (float)h);
  hi[(size_t)row * KD + t] = h;
  lo[(size_t)row * KD + t] = l;

  __shared__ float red[128];
  red[t] = v * v;
  __syncthreads();
  for (int s = 64; s > 0; s >>= 1) {
    if (t < s) red[t] += red[t + s];
    __syncthreads();
  }
  if (t == 0) nrm[row] = red[0];
}

// ---------------------------------------------------------------------------
// Main: one 128x128 superblock of K(A,B) per workgroup; fused exp + reduce.
__global__ __launch_bounds__(256) void mmd_tile_kernel(
    const __bf16* __restrict__ Ahi, const __bf16* __restrict__ Alo,
    const __bf16* __restrict__ Bhi, const __bf16* __restrict__ Blo,
    const float* __restrict__ An, const float* __restrict__ Bn,
    double* __restrict__ accum, int symmetric)
{
  const int bx = blockIdx.x, by = blockIdx.y;
  if (symmetric && bx < by) return;           // upper triangle only (whole block exits)

  const int mb0 = by * BLK;                   // global row base (i)
  const int nb0 = bx * BLK;                   // global col base (j)
  const int tid  = threadIdx.x;
  const int wave = tid >> 5;
  const int lane = tid & 31;
  const int lm = lane & 15;                   // row/col within a 16-tile
  const int h  = lane >> 4;                   // half-wave select
  const int wm = wave >> 1;                   // 0..3 : M strip of 32 rows
  const int wn = wave & 1;                    // 0..1 : N strip of 64 cols

  // Double-buffered chunk storage: [buf][hi/lo][row*LSTR + k]  (~82 KB total;
  // gfx1250 allows up to 320 KB per workgroup -> 3 workgroups/WGP).
  __shared__ __bf16 sA[2][2][BLK * LSTR];
  __shared__ __bf16 sB[2][2][BLK * LSTR];
  __shared__ float  sAn[BLK], sBn[BLK];
  __shared__ float  sred[256];

  if (tid < BLK) sAn[tid] = An[mb0 + tid];
  else           sBn[tid - BLK] = Bn[nb0 + tid - BLK];

  const v8f vzero = {0.f, 0.f, 0.f, 0.f, 0.f, 0.f, 0.f, 0.f};
  v8f acc[2][4];
#pragma unroll
  for (int mt = 0; mt < 2; ++mt)
#pragma unroll
    for (int nt = 0; nt < 4; ++nt) acc[mt][nt] = vzero;

  // Prologue: kick off chunk 0 into buffer 0.
  stage_chunk(Ahi, Alo, Bhi, Blo, &sA[0][0][0], &sA[0][1][0],
              &sB[0][0][0], &sB[0][1][0], mb0, nb0, 0, tid);

  // Pipeline: wait(chunk kc) -> barrier -> issue(chunk kc+1, other buffer)
  //           -> WMMA(chunk kc).  The async DMA of kc+1 runs under the
  //           matrix math of kc; buffer (kc+1)&1 was last read in kc-1,
  //           which every wave finished before this barrier.
#pragma unroll
  for (int kc = 0; kc < NK; ++kc) {
    wait_async_all();
    __syncthreads();
    const int cur = kc & 1, nxt = cur ^ 1;
    if (kc + 1 < NK)
      stage_chunk(Ahi, Alo, Bhi, Blo, &sA[nxt][0][0], &sA[nxt][1][0],
                  &sB[nxt][0][0], &sB[nxt][1][0], mb0, nb0, kc + 1, tid);
    compute_chunk(&sA[cur][0][0], &sA[cur][1][0], &sB[cur][0][0], &sB[cur][1][0],
                  wm, wn, lm, h, acc);
  }

  // Fused epilogue: d2 -> clamp -> exp -> sum. C/D layout: VGPR r holds
  // row (r + 8h) of the tile, column = lane&15.
  float wsum = 0.f;
#pragma unroll
  for (int mt = 0; mt < 2; ++mt) {
#pragma unroll
    for (int nt = 0; nt < 4; ++nt) {
      const int jloc = wn * 64 + nt * 16 + lm;
      const int gj   = nb0 + jloc;
      const float ynv = sBn[jloc];
#pragma unroll
      for (int r = 0; r < 8; ++r) {
        const int iloc = wm * 32 + mt * 16 + 8 * h + r;
        const int gi   = mb0 + iloc;
        const float xnv = sAn[iloc];
        float d2 = fmaf(-2.f, acc[mt][nt][r], xnv + ynv);
        d2 = fmaxf(d2, 0.f);
        const float e = __builtin_amdgcn_exp2f(-NEG_HALF_LOG2E * d2);
        if (!(symmetric && gi == gj)) wsum += e;   // exact diagonal exclusion
      }
    }
  }
  if (symmetric && bx > by) wsum *= 2.f;           // off-diagonal block counted twice

  sred[tid] = wsum;
  __syncthreads();
  for (int s = 128; s > 0; s >>= 1) {
    if (tid < s) sred[tid] += sred[tid + s];
    __syncthreads();
  }
  if (tid == 0) atomicAdd(accum, (double)sred[0]); // global_atomic_add_f64
}

// ---------------------------------------------------------------------------
// Final scalar combine (reference scaling).
__global__ void mmd_final_kernel(const double* __restrict__ accs,
                                 float* __restrict__ out, int N)
{
  if (threadIdx.x == 0 && blockIdx.x == 0) {
    const double n = (double)N;
    const double scale = (n - 1.0) / n;
    const double mmd = (accs[0] + accs[1]) * scale - 2.0 * accs[2] / (n * n);
    out[0] = (float)mmd;
  }
}

// ---------------------------------------------------------------------------
extern "C" void kernel_launch(void* const* d_in, const int* in_sizes, int n_in,
                              void* d_out, int out_size, void* d_ws, size_t ws_size,
                              hipStream_t stream) {
  const float* z  = (const float*)d_in[0];
  const float* zp = (const float*)d_in[1];
  const int N = in_sizes[0] / KD;             // 8192

  // Workspace layout (~8.5 MB): 4x bf16 matrices, 2x norm vectors, 3 doubles.
  char* ws = (char*)d_ws;
  const size_t mat = (size_t)N * KD * sizeof(__bf16);
  __bf16* xhi = (__bf16*)(ws + 0 * mat);
  __bf16* xlo = (__bf16*)(ws + 1 * mat);
  __bf16* yhi = (__bf16*)(ws + 2 * mat);
  __bf16* ylo = (__bf16*)(ws + 3 * mat);
  float*  xn  = (float*)(ws + 4 * mat);
  float*  yn  = (float*)(ws + 4 * mat + (size_t)N * sizeof(float));
  double* accs = (double*)(ws + 4 * mat + 2 * (size_t)N * sizeof(float));

  mmd_prep_kernel<<<dim3(N, 2), dim3(128), 0, stream>>>(
      z, zp, xhi, xlo, yhi, ylo, xn, yn, accs);

  const int nb = N / BLK;                     // 64
  dim3 g(nb, nb), b(256);
  mmd_tile_kernel<<<g, b, 0, stream>>>(xhi, xlo, xhi, xlo, xn, xn, accs + 0, 1); // Kxx
  mmd_tile_kernel<<<g, b, 0, stream>>>(yhi, ylo, yhi, ylo, yn, yn, accs + 1, 1); // Kyy
  mmd_tile_kernel<<<g, b, 0, stream>>>(xhi, xlo, yhi, ylo, xn, yn, accs + 2, 0); // Kxy

  mmd_final_kernel<<<1, 1, 0, stream>>>(accs, (float*)d_out, N);
}
